// InteractionNetwork_8727373545621
// MI455X (gfx1250) — matrix operand order repaired and checked by reading, not compiled, not run on hardware
//
#include <hip/hip_runtime.h>
#include <hip/hip_bf16.h>

typedef __bf16 bf16_t;
typedef __attribute__((ext_vector_type(16))) __bf16 v16bf;
typedef __attribute__((ext_vector_type(8)))  __bf16 bf16x8;
typedef __attribute__((ext_vector_type(4)))  __bf16 bf16x4;
typedef __attribute__((ext_vector_type(8)))  float  v8f;

__device__ __forceinline__ bf16_t f2bf(float x) {
  union { float f; unsigned u; } cv; cv.f = x;
  unsigned u = cv.u;
  u += 0x7FFFu + ((u >> 16) & 1u);            // round-to-nearest-even
  unsigned short h = (unsigned short)(u >> 16);
  union { unsigned short s; bf16_t b; } r; r.s = h;
  return r.b;
}

__device__ __forceinline__ void zero_acc(v8f acc[8]) {
#pragma unroll
  for (int i = 0; i < 8; ++i)
#pragma unroll
    for (int j = 0; j < 8; ++j) acc[i][j] = 0.0f;
}

// One 16xK @ Kx128 GEMM step set: A fragments from LDS (ISA 16-bit A layout),
// B fragments pre-packed in global memory, D accumulated in 8 f32 WMMA tiles.
template <int NUMK>
__device__ __forceinline__ void wmma_tile(const bf16_t* As, int astride,
                                          const bf16_t* Wp, int lane, v8f acc[8]) {
  const int half = lane >> 4;
  const int row  = lane & 15;
#pragma unroll
  for (int kb = 0; kb < NUMK; ++kb) {
    const int kk = kb * 32;
    bf16x8 lo = *reinterpret_cast<const bf16x8*>(As + row * astride + kk + 8 * half);
    bf16x8 hi = *reinterpret_cast<const bf16x8*>(As + row * astride + kk + 16 + 8 * half);
    v16bf a;
#pragma unroll
    for (int i = 0; i < 8; ++i) { a[i] = lo[i]; a[i + 8] = hi[i]; }
#pragma unroll
    for (int nt = 0; nt < 8; ++nt) {
      v16bf b = *reinterpret_cast<const v16bf*>(
          Wp + (((size_t)(nt * NUMK + kb)) * 32 + (size_t)lane) * 16);
      acc[nt] = __builtin_amdgcn_wmma_f32_16x16x32_bf16(
          false, a, false, b, (short)0, acc[nt], false, false);
    }
  }
}

// D (f32, C/D layout: col=lane&15, row=8*half+r) + bias, ReLU, bf16 -> LDS (A layout row-major)
__device__ __forceinline__ void bias_relu_to_lds(const v8f acc[8], const float* bias,
                                                 bf16_t* Hs, int hstride, int lane) {
  const int half = lane >> 4, colL = lane & 15;
#pragma unroll
  for (int nt = 0; nt < 8; ++nt) {
    const int col = nt * 16 + colL;
    const float bb = bias[col];
#pragma unroll
    for (int r = 0; r < 8; ++r) {
      float v = acc[nt][r] + bb;
      v = fmaxf(v, 0.0f);
      Hs[(half * 8 + r) * hstride + col] = f2bf(v);
    }
  }
}

__device__ __forceinline__ void bias_to_f32lds(const v8f acc[8], const float* bias,
                                               float* Sf, int lane) {
  const int half = lane >> 4, colL = lane & 15;
#pragma unroll
  for (int nt = 0; nt < 8; ++nt) {
    const int col = nt * 16 + colL;
    const float bb = bias[col];
#pragma unroll
    for (int r = 0; r < 8; ++r)
      Sf[(half * 8 + r) * 132 + col] = acc[nt][r] + bb;
  }
}

// Pre-pack a (Kdim x 128) f32 weight matrix into WMMA B-fragment order (bf16).
// B 16-bit layout: j<8 -> K=kk+8*half+j, j>=8 -> K=kk+16+8*half+(j-8) ; N = nt*16 + (lane&15)
__global__ void pack_w_kernel(const float* __restrict__ W, bf16_t* __restrict__ out,
                              int Kdim, int numK) {
  const int tid = blockIdx.x * blockDim.x + threadIdx.x;
  if (tid >= Kdim * 128) return;
  const int j    = tid & 15;
  const int lane = (tid >> 4) & 31;
  const int frag = tid >> 9;                 // 512 elements per fragment
  const int kb   = frag % numK;
  const int nt   = frag / numK;
  const int half = lane >> 4;
  const int k    = kb * 32 + (j < 8 ? 8 * half + j : 16 + 8 * half + (j - 8));
  const int n    = nt * 16 + (lane & 15);
  out[tid] = f2bf(W[(size_t)k * 128 + n]);
}

__global__ void zero_kernel(float* p, long n) {
  long i = (long)blockIdx.x * blockDim.x + threadIdx.x;
  if (i < n) p[i] = 0.0f;
}

// ---- Edge pass: msg = LN(MLP([x_i|x_j|e])); atomicAdd(agg[dst], e + msg) ----
__global__ __launch_bounds__(64) void in_edge_kernel(
    const float* __restrict__ node_x, const float* __restrict__ edge_attr,
    const int* __restrict__ eidx,
    const bf16_t* __restrict__ w0p, const bf16_t* __restrict__ w1p, const bf16_t* __restrict__ w2p,
    const float* __restrict__ b0, const float* __restrict__ b1, const float* __restrict__ b2,
    const float* __restrict__ ln_g, const float* __restrict__ ln_b,
    float* __restrict__ agg, int E) {
  extern __shared__ char smem[];
  const int lane = threadIdx.x & 31;
  const int wave = threadIdx.x >> 5;
  char* wb = smem + (size_t)wave * (16 * 392 * 2 + 16 * 136 * 2);
  bf16_t* As = reinterpret_cast<bf16_t*>(wb);            // 16 x 392 bf16 (padded)
  bf16_t* Hs = reinterpret_cast<bf16_t*>(wb + 16 * 392 * 2); // 16 x 136 bf16
  float*  Sf = reinterpret_cast<float*>(wb);             // 16 x 132 f32, aliases As

  const long tile = (long)blockIdx.x * 2 + wave;
  const long e0 = tile * 16;

  // Stage A0 = [x_i | x_j | edge_attr] (16 x 384) as bf16 into LDS.
  for (int r = 0; r < 16; ++r) {
    const long e = e0 + r;
    const long ec = (e < E) ? e : (long)(E - 1);
    const int di = eidx[(size_t)E + (size_t)ec];
    const int si = eidx[(size_t)ec];
    const float* rp0 = node_x + (size_t)di * 128;
    const float* rp1 = node_x + (size_t)si * 128;
    const float* rp2 = edge_attr + (size_t)ec * 128;
#pragma unroll
    for (int t = 0; t < 3; ++t) {
      const int col = (lane + 32 * t) * 4;               // 0..380, never crosses seg
      const int seg = col >> 7, within = col & 127;
      const float* sp = (seg == 0 ? rp0 : (seg == 1 ? rp1 : rp2)) + within;
      float4 v = *reinterpret_cast<const float4*>(sp);
      if (e >= E) v = make_float4(0.f, 0.f, 0.f, 0.f);
      bf16x4 pk; pk[0] = f2bf(v.x); pk[1] = f2bf(v.y); pk[2] = f2bf(v.z); pk[3] = f2bf(v.w);
      *reinterpret_cast<bf16x4*>(&As[r * 392 + col]) = pk;
    }
  }
  __syncthreads();

  v8f acc[8];
  zero_acc(acc);
  wmma_tile<12>(As, 392, w0p, lane, acc);                // 384 -> 128
  bias_relu_to_lds(acc, b0, Hs, 136, lane);
  zero_acc(acc);
  wmma_tile<4>(Hs, 136, w1p, lane, acc);                 // 128 -> 128
  bias_relu_to_lds(acc, b1, Hs, 136, lane);
  zero_acc(acc);
  wmma_tile<4>(Hs, 136, w2p, lane, acc);                 // 128 -> 128 (no act)
  bias_to_f32lds(acc, b2, Sf, lane);                     // As region is dead now

  // LayerNorm over D=128 per row: 2 lanes per row, 64 cols each.
  const int row = lane & 15;
  const int cbase = (lane >> 4) * 64;
  float s = 0.f, s2 = 0.f;
#pragma unroll
  for (int i = 0; i < 64; i += 4) {
    float4 v = *reinterpret_cast<const float4*>(&Sf[row * 132 + cbase + i]);
    s  += v.x + v.y + v.z + v.w;
    s2 += v.x * v.x + v.y * v.y + v.z * v.z + v.w * v.w;
  }
  s  += __shfl_xor(s, 16);
  s2 += __shfl_xor(s2, 16);
  const float mean = s * (1.0f / 128.0f);
  const float var  = s2 * (1.0f / 128.0f) - mean * mean;
  const float rstd = rsqrtf(var + 1e-5f);

  const long e = e0 + row;
  if (e < E) {
    const int di = eidx[(size_t)E + (size_t)e];
    const float* ea = edge_attr + (size_t)e * 128;
    float* ag = agg + (size_t)di * 128;
#pragma unroll 4
    for (int i = 0; i < 64; ++i) {
      const int c = cbase + i;
      const float msg = (Sf[row * 132 + c] - mean) * rstd * ln_g[c] + ln_b[c];
      atomicAdd(&ag[c], ea[c] + msg);                    // global_atomic_add_f32
    }
  }
}

// ---- Node pass: out = x + LN(MLP([x | agg])) ----
__global__ __launch_bounds__(64) void in_node_kernel(
    const float* __restrict__ node_x, const float* __restrict__ agg,
    const bf16_t* __restrict__ w0p, const bf16_t* __restrict__ w1p, const bf16_t* __restrict__ w2p,
    const float* __restrict__ b0, const float* __restrict__ b1, const float* __restrict__ b2,
    const float* __restrict__ ln_g, const float* __restrict__ ln_b,
    float* __restrict__ out, int N) {
  extern __shared__ char smem[];
  const int lane = threadIdx.x & 31;
  const int wave = threadIdx.x >> 5;
  char* wb = smem + (size_t)wave * (16 * 264 * 2 + 16 * 136 * 2);
  bf16_t* As = reinterpret_cast<bf16_t*>(wb);            // 16 x 264 bf16
  bf16_t* Hs = reinterpret_cast<bf16_t*>(wb + 16 * 264 * 2);
  float*  Sf = reinterpret_cast<float*>(wb);             // 16 x 132 f32 alias

  const long tile = (long)blockIdx.x * 2 + wave;
  const long n0 = tile * 16;

  for (int r = 0; r < 16; ++r) {
    const long n = n0 + r;
    const long nc = (n < N) ? n : (long)(N - 1);
    const float* rp0 = node_x + (size_t)nc * 128;
    const float* rp1 = agg    + (size_t)nc * 128;
#pragma unroll
    for (int t = 0; t < 2; ++t) {
      const int col = (lane + 32 * t) * 4;               // 0..252
      const int seg = col >> 7, within = col & 127;
      const float* sp = (seg == 0 ? rp0 : rp1) + within;
      float4 v = *reinterpret_cast<const float4*>(sp);
      if (n >= N) v = make_float4(0.f, 0.f, 0.f, 0.f);
      bf16x4 pk; pk[0] = f2bf(v.x); pk[1] = f2bf(v.y); pk[2] = f2bf(v.z); pk[3] = f2bf(v.w);
      *reinterpret_cast<bf16x4*>(&As[r * 264 + col]) = pk;
    }
  }
  __syncthreads();

  v8f acc[8];
  zero_acc(acc);
  wmma_tile<8>(As, 264, w0p, lane, acc);                 // 256 -> 128
  bias_relu_to_lds(acc, b0, Hs, 136, lane);
  zero_acc(acc);
  wmma_tile<4>(Hs, 136, w1p, lane, acc);
  bias_relu_to_lds(acc, b1, Hs, 136, lane);
  zero_acc(acc);
  wmma_tile<4>(Hs, 136, w2p, lane, acc);
  bias_to_f32lds(acc, b2, Sf, lane);

  const int row = lane & 15;
  const int cbase = (lane >> 4) * 64;
  float s = 0.f, s2 = 0.f;
#pragma unroll
  for (int i = 0; i < 64; i += 4) {
    float4 v = *reinterpret_cast<const float4*>(&Sf[row * 132 + cbase + i]);
    s  += v.x + v.y + v.z + v.w;
    s2 += v.x * v.x + v.y * v.y + v.z * v.z + v.w * v.w;
  }
  s  += __shfl_xor(s, 16);
  s2 += __shfl_xor(s2, 16);
  const float mean = s * (1.0f / 128.0f);
  const float var  = s2 * (1.0f / 128.0f) - mean * mean;
  const float rstd = rsqrtf(var + 1e-5f);

  const long n = n0 + row;
  if (n < N) {
    const float* nx = node_x + (size_t)n * 128;
    float* op = out + (size_t)n * 128;
#pragma unroll 4
    for (int i = 0; i < 64; ++i) {
      const int c = cbase + i;
      const float u = (Sf[row * 132 + c] - mean) * rstd * ln_g[c] + ln_b[c];
      op[c] = nx[c] + u;
    }
  }
}

extern "C" void kernel_launch(void* const* d_in, const int* in_sizes, int n_in,
                              void* d_out, int out_size, void* d_ws, size_t ws_size,
                              hipStream_t stream) {
  const float* node_x    = (const float*)d_in[0];
  const float* edge_attr = (const float*)d_in[1];
  const int*   eidx      = (const int*)d_in[2];
  const float* msg_w0 = (const float*)d_in[3];
  const float* msg_b0 = (const float*)d_in[4];
  const float* msg_w1 = (const float*)d_in[5];
  const float* msg_b1 = (const float*)d_in[6];
  const float* msg_w2 = (const float*)d_in[7];
  const float* msg_b2 = (const float*)d_in[8];
  const float* msg_g  = (const float*)d_in[9];
  const float* msg_be = (const float*)d_in[10];
  const float* upd_w0 = (const float*)d_in[11];
  const float* upd_b0 = (const float*)d_in[12];
  const float* upd_w1 = (const float*)d_in[13];
  const float* upd_b1 = (const float*)d_in[14];
  const float* upd_w2 = (const float*)d_in[15];
  const float* upd_b2 = (const float*)d_in[16];
  const float* upd_g  = (const float*)d_in[17];
  const float* upd_be = (const float*)d_in[18];

  const int N = in_sizes[0] / 128;
  const int E = in_sizes[1] / 128;

  // Workspace: f32 agg [N,128] then bf16 fragment-packed weights.
  char* ws = (char*)d_ws;
  float* agg = (float*)ws;
  size_t off = (((size_t)N * 128 * sizeof(float)) + 255) & ~(size_t)255;
  bf16_t* mw0 = (bf16_t*)(ws + off); off += (size_t)384 * 128 * 2;
  bf16_t* mw1 = (bf16_t*)(ws + off); off += (size_t)128 * 128 * 2;
  bf16_t* mw2 = (bf16_t*)(ws + off); off += (size_t)128 * 128 * 2;
  bf16_t* uw0 = (bf16_t*)(ws + off); off += (size_t)256 * 128 * 2;
  bf16_t* uw1 = (bf16_t*)(ws + off); off += (size_t)128 * 128 * 2;
  bf16_t* uw2 = (bf16_t*)(ws + off); off += (size_t)128 * 128 * 2;

  const long aggN = (long)N * 128;
  zero_kernel<<<(int)((aggN + 255) / 256), 256, 0, stream>>>(agg, aggN);

  pack_w_kernel<<<(384 * 128 + 255) / 256, 256, 0, stream>>>(msg_w0, mw0, 384, 12);
  pack_w_kernel<<<(128 * 128 + 255) / 256, 256, 0, stream>>>(msg_w1, mw1, 128, 4);
  pack_w_kernel<<<(128 * 128 + 255) / 256, 256, 0, stream>>>(msg_w2, mw2, 128, 4);
  pack_w_kernel<<<(256 * 128 + 255) / 256, 256, 0, stream>>>(upd_w0, uw0, 256, 8);
  pack_w_kernel<<<(128 * 128 + 255) / 256, 256, 0, stream>>>(upd_w1, uw1, 128, 4);
  pack_w_kernel<<<(128 * 128 + 255) / 256, 256, 0, stream>>>(upd_w2, uw2, 128, 4);

  const int edgeBlocks = (E + 31) / 32;                  // 2 waves/block, 16 edges/wave
  const size_t edgeShm = 2 * (16 * 392 * 2 + 16 * 136 * 2);
  in_edge_kernel<<<edgeBlocks, 64, edgeShm, stream>>>(
      node_x, edge_attr, eidx, mw0, mw1, mw2,
      msg_b0, msg_b1, msg_b2, msg_g, msg_be, agg, E);

  const int nodeBlocks = (N + 31) / 32;
  const size_t nodeShm = 2 * (16 * 264 * 2 + 16 * 136 * 2);
  in_node_kernel<<<nodeBlocks, 64, nodeShm, stream>>>(
      node_x, agg, uw0, uw1, uw2,
      upd_b0, upd_b1, upd_b2, upd_g, upd_be, (float*)d_out, N);
}
